// ResonancePartialCNN_23338852286674
// MI455X (gfx1250) — matrix operand order repaired and checked by reading, not compiled, not run on hardware
//
#include <hip/hip_runtime.h>
#include <cstdint>

typedef __attribute__((ext_vector_type(16))) _Float16 v16h;
typedef __attribute__((ext_vector_type(8)))  _Float16 v8h;
typedef __attribute__((ext_vector_type(8)))  float    v8f;

// A-fragment (16x32 f16, M x K): lane m = lane%16; contiguous K chunk at
// offset (lane<16 ? 0 : 16), 16 elements.
__device__ __forceinline__ v16h load_a_frag(const _Float16* row, int lane) {
  const int c0 = (lane < 16) ? 0 : 16;
  v8h lo = *(const v8h*)(row + c0);
  v8h hi = *(const v8h*)(row + c0 + 8);
  return __builtin_shufflevector(lo, hi, 0,1,2,3,4,5,6,7,8,9,10,11,12,13,14,15);
}

// ---------------------------------------------------------------------------
// Stage-0 prep: split x into masked data (f16) and clipped mask (f32)
// ---------------------------------------------------------------------------
__global__ void k_extract(const float* __restrict__ x, _Float16* __restrict__ xm,
                          float* __restrict__ mask, int HW) {
  int n = blockIdx.y;
  int i = blockIdx.x * blockDim.x + threadIdx.x;
  if (i >= HW) return;
  float d = x[((long)n * 2 + 0) * HW + i];
  float m = x[((long)n * 2 + 1) * HW + i];
  m = fminf(fmaxf(m, 0.f), 1.f);
  mask[(long)n * HW + i] = m;
  xm[(long)n * HW + i] = (_Float16)(d * m);
}

// Convert conv weights (Co,Ci,3,3) f32 -> f16 [Co][Kpad], zero padded in K
__global__ void k_cvt_w(const float* __restrict__ w, _Float16* __restrict__ wf,
                        int Co, int K, int Kpad) {
  long idx = (long)blockIdx.x * blockDim.x + threadIdx.x;
  if (idx >= (long)Co * Kpad) return;
  int co = (int)(idx / Kpad), k = (int)(idx % Kpad);
  wf[idx] = (k < K) ? (_Float16)w[(long)co * K + k] : (_Float16)0.f;
}

// Elementwise f32 -> f16
__global__ void k_cvt_h(const float* __restrict__ a, _Float16* __restrict__ b, long n) {
  long i = (long)blockIdx.x * blockDim.x + threadIdx.x;
  if (i < n) b[i] = (_Float16)a[i];
}

// ---------------------------------------------------------------------------
// Partial conv as implicit GEMM, WMMA f32_16x16x32_f16.
// Block tile: 64 output channels x 64 contiguous pixels (one image row seg).
// Block = 128 threads (4 waves). Wave w owns co-subtile w (16 rows) and runs
// 4 N-subtiles -> 16 WMMAs per K-chunk per block.
// Weights (contiguous) are staged global->LDS with CDNA5 async-to-LDS b128;
// the im2col B tile is gathered by all threads into LDS.
// ---------------------------------------------------------------------------
__global__ void k_pconv_wmma(const _Float16* __restrict__ xm,
                             const float* __restrict__ mask,
                             const _Float16* __restrict__ wf,
                             const float* __restrict__ bias,
                             float* __restrict__ y,
                             float* __restrict__ mask_next,
                             int Ci, int Co, int H, int Kreal, int Kpad) {
  constexpr int W = 128;                 // invariant across all stages
  __shared__ _Float16 At[64][32];        // weight tile   (4 KB)
  __shared__ _Float16 Bt[64][32];        // im2col tile   (4 KB)

  const int tid  = threadIdx.x;
  const int lane = tid & 31;
  const int wave = tid >> 5;
  const int nb   = blockIdx.z;
  const int pbase = blockIdx.x * 64;     // 64 pixels, single row (64 | 128)
  const int py   = pbase >> 7;
  const int px0  = pbase & (W - 1);
  const int cob  = blockIdx.y * 64;      // block channel base
  const int co0  = cob + wave * 16;      // wave channel base
  const bool covalid = (co0 < Co);

  const unsigned at_base = (unsigned)(uintptr_t)(&At[0][0]);  // LDS byte offset

  v8f acc[4];
  #pragma unroll
  for (int t = 0; t < 4; ++t)
    acc[t] = (v8f){0.f, 0.f, 0.f, 0.f, 0.f, 0.f, 0.f, 0.f};

  for (int kc = 0; kc < Kpad; kc += 32) {
    // ---- async stage of the 64x32 A tile: 256 b128 ops, 2 per thread ----
    #pragma unroll
    for (int j = 0; j < 2; ++j) {
      int q   = j * 128 + tid;                 // b128 index: row = q/4, col16B = q%4
      int row = q >> 2;
      int c8  = q & 3;
      int cog = cob + row;
      if (cog >= Co) cog = Co - 1;             // clamp: stay in-bounds, rows unused
      unsigned long long ga =
          (unsigned long long)(const void*)(wf + (long)cog * Kpad + kc + c8 * 8);
      unsigned lds = at_base + (unsigned)q * 16u;
      asm volatile("global_load_async_to_lds_b128 %0, %1, off"
                   :: "v"(lds), "v"(ga) : "memory");
    }

    // ---- cooperative im2col gather of the 64x32 B tile (16 per thread) ----
    #pragma unroll
    for (int j = 0; j < 16; ++j) {
      int e  = j * 128 + tid;
      int n  = e >> 5;
      int kk = e & 31;
      int k  = kc + kk;
      _Float16 v = (_Float16)0.f;
      if (k < Kreal) {
        int ci = k / 9;
        int r  = k - ci * 9;
        int yy = py + r / 3 - 1;
        int xx = px0 + n + (r % 3) - 1;
        if (yy >= 0 && yy < H && xx >= 0 && xx < W)
          v = xm[(((long)nb * Ci + ci) * H + yy) * W + xx];
      }
      Bt[n][kk] = v;
    }

    asm volatile("s_wait_asynccnt 0" ::: "memory");
    __syncthreads();

    v16h a = load_a_frag(&At[wave * 16 + (lane & 15)][0], lane);
    #pragma unroll
    for (int t = 0; t < 4; ++t) {
      v16h b = *(const v16h*)(&Bt[t * 16 + (lane & 15)][(lane >> 4) * 16]);
      acc[t] = __builtin_amdgcn_wmma_f32_16x16x32_f16(false, a, false, b,
                                                      (short)0, acc[t], false, false);
    }
    __syncthreads();
  }

  // ---- epilogue: partial-conv renormalization per pixel + store ----
  #pragma unroll
  for (int t = 0; t < 4; ++t) {
    const int n  = t * 16 + (lane & 15);
    const int px = px0 + n;
    float msum = 0.f;
    #pragma unroll
    for (int dy = -1; dy <= 1; ++dy)
      #pragma unroll
      for (int dx = -1; dx <= 1; ++dx) {
        int yy = py + dy, xx = px + dx;
        msum += (yy >= 0 && yy < H && xx >= 0 && xx < W)
                    ? mask[((long)nb * H + yy) * W + xx] : 1.0f;
      }
    const float validc  = (float)Ci * msum;
    const bool  novalid = (validc <= 0.f);
    const float scale   = (float)Kreal / fmaxf(validc, 1.0f);

    if (covalid) {
      #pragma unroll
      for (int r = 0; r < 8; ++r) {
        int mm = ((lane >> 4) << 3) + r;       // D layout: m = 8*(lane/16)+r
        int co = co0 + mm;
        float v = novalid ? 0.f : acc[t][r] * scale + bias[co];
        y[(((long)nb * Co + co) * H + py) * W + px] = v;
      }
    }
    if (blockIdx.y == 0 && wave == 0 && lane < 16)
      mask_next[((long)nb * H + py) * W + px] = novalid ? 0.f : 1.f;
  }
}

// ---------------------------------------------------------------------------
// BatchNorm statistics: one block per channel, reduce over N*H*W
// ---------------------------------------------------------------------------
__global__ void k_bn_stats(const float* __restrict__ y, float* __restrict__ mean,
                           float* __restrict__ var, int N, int C, int H, int W) {
  const int c = blockIdx.x;
  const long HW = (long)H * W;
  const long cnt = (long)N * HW;
  float s = 0.f, s2 = 0.f;
  for (long i = threadIdx.x; i < cnt; i += blockDim.x) {
    long n = i / HW, hw = i - n * HW;
    float v = y[(n * C + c) * HW + hw];
    s += v; s2 += v * v;
  }
  __shared__ float sh[256], sh2[256];
  sh[threadIdx.x] = s; sh2[threadIdx.x] = s2;
  __syncthreads();
  for (int off = 128; off > 0; off >>= 1) {
    if ((int)threadIdx.x < off) {
      sh[threadIdx.x]  += sh[threadIdx.x + off];
      sh2[threadIdx.x] += sh2[threadIdx.x + off];
    }
    __syncthreads();
  }
  if (threadIdx.x == 0) {
    float mu = sh[0] / (float)cnt;
    mean[c] = mu;
    var[c]  = sh2[0] / (float)cnt - mu * mu;
  }
}

// Fused BN + ReLU + (2,1) max-pool (halves H)
__global__ void k_bn_relu_pool(const float* __restrict__ y,
                               const float* __restrict__ mean, const float* __restrict__ var,
                               const float* __restrict__ g, const float* __restrict__ be,
                               float* __restrict__ out, int N, int C, int H, int W) {
  const int H2 = H >> 1;
  const long total = (long)N * C * H2 * W;
  long i = (long)blockIdx.x * blockDim.x + threadIdx.x;
  if (i >= total) return;
  int w = (int)(i % W);  long t = i / W;
  int h2 = (int)(t % H2); t /= H2;
  int c = (int)(t % C);   int n = (int)(t / C);
  float mu = mean[c];
  float iv = rsqrtf(var[c] + 1e-5f);
  float ga = g[c], bb = be[c];
  long base = (((long)n * C + c) * H + 2 * h2) * W + w;
  float a0 = fmaxf((y[base]     - mu) * iv * ga + bb, 0.f);
  float a1 = fmaxf((y[base + W] - mu) * iv * ga + bb, 0.f);
  out[i] = fmaxf(a0, a1);
}

// (2,1) max-pool on the mask
__global__ void k_pool_mask(const float* __restrict__ m, float* __restrict__ out,
                            int N, int H, int W) {
  const int H2 = H >> 1;
  const long total = (long)N * H2 * W;
  long i = (long)blockIdx.x * blockDim.x + threadIdx.x;
  if (i >= total) return;
  int w = (int)(i % W);  long t = i / W;
  int h2 = (int)(t % H2); int n = (int)(t / H2);
  long base = ((long)n * H + 2 * h2) * W + w;
  out[i] = fmaxf(m[base], m[base + W]);
}

// xm_next = (f16)(h_pooled * mask_pooled)  (mask broadcast over channels)
__global__ void k_maskmul_cvt(const float* __restrict__ h, const float* __restrict__ m,
                              _Float16* __restrict__ xm, int N, int C, int H, int W) {
  const long total = (long)N * C * H * W;
  long i = (long)blockIdx.x * blockDim.x + threadIdx.x;
  if (i >= total) return;
  int w = (int)(i % W);  long t = i / W;
  int hh = (int)(t % H); t /= H;
  int n = (int)(t / C);
  xm[i] = (_Float16)(h[i] * m[((long)n * H + hh) * W + w]);
}

// Global average pool -> feat (f32 + f16)
__global__ void k_gap(const float* __restrict__ h, float* __restrict__ feat,
                      _Float16* __restrict__ feath, int C, int HW) {
  const int c = blockIdx.x, n = blockIdx.y;
  __shared__ float sh[256];
  float s = 0.f;
  const float* p = h + ((long)n * C + c) * HW;
  for (int i = threadIdx.x; i < HW; i += 256) s += p[i];
  sh[threadIdx.x] = s;
  __syncthreads();
  for (int off = 128; off > 0; off >>= 1) {
    if ((int)threadIdx.x < off) sh[threadIdx.x] += sh[threadIdx.x + off];
    __syncthreads();
  }
  if (threadIdx.x == 0) {
    float v = sh[0] / (float)HW;
    feat[(long)n * C + c]  = v;
    feath[(long)n * C + c] = (_Float16)v;
  }
}

// ---------------------------------------------------------------------------
// FC layer via WMMA: out[16][Nout] = A[16][K] @ W^T, W given (Nout,K) row-major
// (== B^T[n][k], exactly the B-fragment layout). One wave per 16-col tile.
// ---------------------------------------------------------------------------
__global__ void k_fc_wmma(const _Float16* __restrict__ A, const _Float16* __restrict__ Bw,
                          const float* __restrict__ bias, float* __restrict__ out,
                          _Float16* __restrict__ outh, int K, int Nout, int do_relu) {
  const int lane = threadIdx.x & 31;
  const int col = blockIdx.x * 16 + (lane & 15);
  v8f acc = {0.f, 0.f, 0.f, 0.f, 0.f, 0.f, 0.f, 0.f};
  for (int kc = 0; kc < K; kc += 32) {
    v16h a = load_a_frag(A + (long)(lane & 15) * K + kc, lane);
    v16h b = *(const v16h*)(Bw + (long)col * K + kc + ((lane >> 4) * 16));
    acc = __builtin_amdgcn_wmma_f32_16x16x32_f16(false, a, false, b,
                                                 (short)0, acc, false, false);
  }
  const float bv = bias[col];
  #pragma unroll
  for (int r = 0; r < 8; ++r) {
    int mrow = ((lane >> 4) << 3) + r;
    float v = acc[r] + bv;
    if (do_relu) v = fmaxf(v, 0.f);
    out[(long)mrow * Nout + col] = v;
    if (outh) outh[(long)mrow * Nout + col] = (_Float16)v;
  }
}

// Head: out[n] = sigmoid(z[n][:] . hw + hb), one block per batch element
__global__ void k_head(const float* __restrict__ z, const float* __restrict__ hw,
                       const float* __restrict__ hb, float* __restrict__ out) {
  __shared__ float sh[128];
  const int n = blockIdx.x, t = threadIdx.x;
  sh[t] = z[(long)n * 128 + t] * hw[t];
  __syncthreads();
  for (int off = 64; off > 0; off >>= 1) {
    if (t < off) sh[t] += sh[t + off];
    __syncthreads();
  }
  if (t == 0) out[n] = 1.f / (1.f + expf(-(sh[0] + hb[0])));
}

// ---------------------------------------------------------------------------
extern "C" void kernel_launch(void* const* d_in, const int* in_sizes, int n_in,
                              void* d_out, int out_size, void* d_ws, size_t ws_size,
                              hipStream_t stream) {
  (void)in_sizes; (void)n_in; (void)out_size; (void)ws_size;
  const float* x = (const float*)d_in[0];
  const float* Wc[4] = {(const float*)d_in[1],  (const float*)d_in[5],
                        (const float*)d_in[9],  (const float*)d_in[13]};
  const float* Bc[4] = {(const float*)d_in[2],  (const float*)d_in[6],
                        (const float*)d_in[10], (const float*)d_in[14]};
  const float* Gc[4] = {(const float*)d_in[3],  (const float*)d_in[7],
                        (const float*)d_in[11], (const float*)d_in[15]};
  const float* Ec[4] = {(const float*)d_in[4],  (const float*)d_in[8],
                        (const float*)d_in[12], (const float*)d_in[16]};
  const float* fw1 = (const float*)d_in[17];
  const float* fb1 = (const float*)d_in[18];
  const float* fw2 = (const float*)d_in[19];
  const float* fb2 = (const float*)d_in[20];
  const float* hwp = (const float*)d_in[21];
  const float* hbp = (const float*)d_in[22];

  // bump allocator on workspace
  char* wsb = (char*)d_ws;
  size_t off = 0;
  auto alloc = [&](size_t bytes) -> void* {
    void* p = wsb + off;
    off += (bytes + 255) & ~(size_t)255;
    return p;
  };
  const int N = 16, Wd = 128;
  const int CiA[4] = {1, 80, 160, 320};
  const int CoA[4] = {80, 160, 320, 640};
  const int HA[4]  = {256, 128, 64, 32};

  _Float16* xm     = (_Float16*)alloc((size_t)20971520 * 2); // max N*Ci*H*W f16
  float*    mask_a = (float*)   alloc((size_t)524288  * 4);  // N*H*W f32 (cur)
  float*    mask_b = (float*)   alloc((size_t)524288  * 4);  // conv-res mask_next
  float*    yraw   = (float*)   alloc((size_t)41943040 * 4); // max N*Co*H*W f32
  float*    hpool  = (float*)   alloc((size_t)20971520 * 4); // pooled activations
  _Float16* wf     = (_Float16*)alloc((size_t)640 * 2880 * 2);
  float*    meanb  = (float*)   alloc(640 * 4);
  float*    varb   = (float*)   alloc(640 * 4);
  float*    feat   = (float*)   alloc(16 * 640 * 4);
  _Float16* feath  = (_Float16*)alloc(16 * 640 * 2);
  _Float16* fw1h   = (_Float16*)alloc(256 * 640 * 2);
  _Float16* fw2h   = (_Float16*)alloc(128 * 256 * 2);
  float*    z1     = (float*)   alloc(16 * 256 * 4);
  _Float16* z1h    = (_Float16*)alloc(16 * 256 * 2);
  float*    z2     = (float*)   alloc(16 * 128 * 4);
  _Float16* z2h    = (_Float16*)alloc(16 * 128 * 2);

  // stage 0 input prep
  {
    int HW = 256 * 128;
    k_extract<<<dim3((HW + 255) / 256, N), 256, 0, stream>>>(x, xm, mask_a, HW);
  }

  for (int s = 0; s < 4; ++s) {
    const int Ci = CiA[s], Co = CoA[s], H = HA[s];
    const int Kreal = Ci * 9;
    const int Kpad  = ((Kreal + 31) / 32) * 32;

    { // weight conversion
      long tot = (long)Co * Kpad;
      k_cvt_w<<<(unsigned)((tot + 255) / 256), 256, 0, stream>>>(Wc[s], wf, Co, Kreal, Kpad);
    }
    { // partial conv (WMMA implicit GEMM, 64x64 block tile)
      dim3 grid((H * Wd) / 64, (Co + 63) / 64, N);
      k_pconv_wmma<<<grid, 128, 0, stream>>>(xm, mask_a, wf, Bc[s], yraw, mask_b,
                                             Ci, Co, H, Kreal, Kpad);
    }
    k_bn_stats<<<Co, 256, 0, stream>>>(yraw, meanb, varb, N, Co, H, Wd);
    {
      long tot = (long)N * Co * (H / 2) * Wd;
      k_bn_relu_pool<<<(unsigned)((tot + 255) / 256), 256, 0, stream>>>(
          yraw, meanb, varb, Gc[s], Ec[s], hpool, N, Co, H, Wd);
    }
    {
      long tot = (long)N * (H / 2) * Wd;
      k_pool_mask<<<(unsigned)((tot + 255) / 256), 256, 0, stream>>>(mask_b, mask_a, N, H, Wd);
    }
    if (s < 3) {
      long tot = (long)N * Co * (H / 2) * Wd;
      k_maskmul_cvt<<<(unsigned)((tot + 255) / 256), 256, 0, stream>>>(
          hpool, mask_a, xm, N, Co, H / 2, Wd);
    }
  }

  // GAP over 16x128 spatial -> feat[16][640]
  k_gap<<<dim3(640, N), 256, 0, stream>>>(hpool, feat, feath, 640, 16 * 128);

  // MLP head (WMMA)
  k_cvt_h<<<(256 * 640 + 255) / 256, 256, 0, stream>>>(fw1, fw1h, 256 * 640);
  k_cvt_h<<<(128 * 256 + 255) / 256, 256, 0, stream>>>(fw2, fw2h, 128 * 256);
  k_fc_wmma<<<256 / 16, 32, 0, stream>>>(feath, fw1h, fb1, z1, z1h, 640, 256, 1);
  k_fc_wmma<<<128 / 16, 32, 0, stream>>>(z1h, fw2h, fb2, z2, z2h, 256, 128, 1);
  k_head<<<16, 128, 0, stream>>>(z2, hwp, hbp, (float*)d_out);
}
